// AttentionCropBlock_64364379898301
// MI455X (gfx1250) — matrix operand order repaired and checked by reading, not compiled, not run on hardware
//
#include <hip/hip_runtime.h>

// ---------- types ----------
typedef __attribute__((ext_vector_type(16))) __bf16 v16bf;
typedef __attribute__((ext_vector_type(8)))  float  v8f;

union FragBF { uint4 u[2]; v16bf v; };

// ---------- helpers ----------
__device__ __forceinline__ unsigned short f2bf(float f) {
    union { float f; unsigned u; } a; a.f = f;
    unsigned u = a.u;
    unsigned r = (u + 0x7fffu + ((u >> 16) & 1u)) >> 16;   // round-to-nearest-even
    return (unsigned short)r;
}
__device__ __forceinline__ unsigned fkey(float f) {        // monotonic float->uint
    int i = __float_as_int(f);
    return (i >= 0) ? ((unsigned)i | 0x80000000u) : ~(unsigned)i;
}
__device__ __forceinline__ float funkey(unsigned k) {
    int i = (k & 0x80000000u) ? (int)(k & 0x7fffffffu) : ~(int)k;
    return __int_as_float(i);
}

// ---------- prep: BN scale/shift ----------
__global__ void prep_bn(const float* g1,const float* b1,const float* m1,const float* v1,
                        const float* g2,const float* b2,const float* m2,const float* v2,
                        const float* g3,const float* b3,const float* m3,const float* v3,
                        float* scaleAll, float* shiftAll) {
    int i = threadIdx.x; if (i >= 192) return;
    int set = i >> 6, j = i & 63;
    const float *g, *b, *m, *v;
    if (set == 0)      { g=g1; b=b1; m=m1; v=v1; }
    else if (set == 1) { g=g2; b=b2; m=m2; v=v2; }
    else               { g=g3; b=b3; m=m3; v=v3; }
    float s = g[j] * rsqrtf(v[j] + 1e-5f);
    scaleAll[i] = s;
    shiftAll[i] = b[j] - m[j] * s;
}

// ---------- prep: conv1 weights (fp32, scale folded) ----------
__global__ void prep_w1(const float* __restrict__ w1, const float* __restrict__ scale,
                        float* __restrict__ w1s) {
    int idx = blockIdx.x * 256 + threadIdx.x;
    if (idx >= 64 * 27) return;
    int oc = idx / 27;
    w1s[idx] = w1[idx] * scale[oc];
}

// ---------- prep: 64->64 weights swizzled into WMMA A-fragment layout, bf16 ----------
// wA[mt][ks][lane][e] with K(lane,e) = 16*(lane/16) + e (contiguous k order), k = tap*64 + ic
__global__ void prep_wA(const float* __restrict__ w, const float* __restrict__ scale,
                        unsigned short* __restrict__ wA) {
    int idx = blockIdx.x * 256 + threadIdx.x;
    if (idx >= 4 * 18 * 32 * 16) return;
    int e    = idx & 15;
    int lane = (idx >> 4) & 31;
    int ks   = (idx >> 9) % 18;
    int mt   = idx / (18 * 32 * 16);
    int K  = e + (lane >= 16 ? 16 : 0);
    int oc = mt * 16 + (lane & 15);
    int kg = ks * 32 + K;               // global K in [0,576)
    int ic = kg & 63, tap = kg >> 6;    // tap = dy*3+dx
    float val = w[((size_t)oc * 64 + ic) * 9 + tap] * scale[oc];
    wA[idx] = f2bf(val);
}

// ---------- conv1: 3->64, BN folded, ReLU, NCHW fp32 -> NHWC bf16 ----------
__global__ __launch_bounds__(256)
void conv1_kernel(const float* __restrict__ x, const float* __restrict__ w1s,
                  const float* __restrict__ shift, unsigned short* __restrict__ h1) {
    __shared__ float lw[64 * 27];
    int tid = threadIdx.x;
    for (int i = tid; i < 64 * 27; i += 256) lw[i] = w1s[i];
    __syncthreads();

    size_t gid = (size_t)blockIdx.x * 256 + tid;
    size_t p = gid >> 2;                 // pixel
    int q = (int)(gid & 3);              // oc group of 16
    if (p >= (size_t)8 * 384 * 384) return;
    int xw = (int)(p % 384); size_t t = p / 384;
    int yh = (int)(t % 384); int b = (int)(t / 384);
    int ocb = q * 16;

    float acc[16];
    #pragma unroll
    for (int o = 0; o < 16; ++o) acc[o] = shift[ocb + o];

    for (int ic = 0; ic < 3; ++ic)
        for (int dy = 0; dy < 3; ++dy) {
            int gy = yh + dy - 1;
            if (gy < 0 || gy >= 384) continue;
            for (int dx = 0; dx < 3; ++dx) {
                int gx = xw + dx - 1;
                if (gx < 0 || gx >= 384) continue;
                float xv = x[(((size_t)b * 3 + ic) * 384 + gy) * 384 + gx];
                int k = ic * 9 + dy * 3 + dx;
                #pragma unroll
                for (int o = 0; o < 16; ++o) acc[o] += xv * lw[(ocb + o) * 27 + k];
            }
        }

    union { unsigned short s[16]; uint4 u[2]; } pk;
    #pragma unroll
    for (int o = 0; o < 16; ++o) { float v = acc[o] > 0.f ? acc[o] : 0.f; pk.s[o] = f2bf(v); }
    uint4* dst = reinterpret_cast<uint4*>(h1 + (((size_t)(b * 384 + yh) * 384 + xw) * 64 + ocb));
    dst[0] = pk.u[0]; dst[1] = pk.u[1];
}

// ---------- 64->64 3x3 conv via WMMA implicit GEMM ----------
// Workgroup: 16x16 pixel tile, 8 waves; each wave owns 2 M-tiles x 4 N-tiles.
// MODE 0: +shift, ReLU, store NHWC bf16.  MODE 1: +shift, channel-max -> sigmoid -> sa.
template<int MODE>
__global__ __launch_bounds__(256)
void conv64_wmma(const unsigned short* __restrict__ in,   // [8][384][384][64] bf16
                 const unsigned short* __restrict__ wA,   // [4][18][32][16] bf16 (A-frag layout)
                 const float* __restrict__ shift,         // [64]
                 unsigned short* __restrict__ out,        // MODE 0
                 float* __restrict__ sa) {                // MODE 1
    __shared__ __align__(32) unsigned short lds_in[18 * 18 * 64];  // 41.5 KB halo tile
    __shared__ unsigned lds_max[16 * 16];

    const int tid = threadIdx.x;
    const int x0 = blockIdx.x * 16, y0 = blockIdx.y * 16, b = blockIdx.z;

    // cooperative halo load: 18x18 pixels x 64ch, uint4 = 8 bf16 at a time
    for (int idx = tid; idx < 18 * 18 * 8; idx += 256) {
        int q = idx & 7, pix = idx >> 3;
        int ry = pix / 18, rx = pix % 18;
        int gy = y0 + ry - 1, gx = x0 + rx - 1;
        uint4 v = make_uint4(0, 0, 0, 0);
        if (gy >= 0 && gy < 384 && gx >= 0 && gx < 384)
            v = *reinterpret_cast<const uint4*>(in + (((size_t)(b * 384 + gy) * 384 + gx) * 64 + q * 8));
        *reinterpret_cast<uint4*>(&lds_in[pix * 64 + q * 8]) = v;
    }
    if (MODE == 1) lds_max[tid] = 0u;
    __syncthreads();

    const int wave = tid >> 5, lane = tid & 31;
    const int mpair = wave & 1;             // M-tiles 2*mpair, 2*mpair+1
    const int m0 = mpair * 2;
    const int nbase = (wave >> 1) * 4;      // 4 N-tile rows per wave
    const int col = lane & 15;
    const int hi  = (lane & 16) ? 1 : 0;

    v8f acc[2][4];
    #pragma unroll
    for (int j = 0; j < 2; ++j)
        #pragma unroll
        for (int n = 0; n < 4; ++n)
            #pragma unroll
            for (int r = 0; r < 8; ++r) acc[j][n][r] = 0.f;

    for (int ks = 0; ks < 18; ++ks) {       // K = 576 = 18 x 32
        // A fragments (2 M-tiles), straight from global (L2/L1-hot, shared by all blocks)
        FragBF a[2];
        #pragma unroll
        for (int j = 0; j < 2; ++j) {
            const uint4* ap = reinterpret_cast<const uint4*>(
                wA + ((size_t)(((m0 + j) * 18 + ks) * 32 + lane) * 16));
            a[j].u[0] = ap[0]; a[j].u[1] = ap[1];
        }
        int tap = ks >> 1, dy = tap / 3, dx = tap % 3;
        int chb = ((ks & 1) << 5) + (hi << 4);
        int rxc = col + dx;
        // batch all B fragment loads (distinct regs -> partial dscnt waits, DS/WMMA overlap)
        FragBF bf[4];
        #pragma unroll
        for (int n = 0; n < 4; ++n) {
            int ry = nbase + n + dy;
            const uint4* bp = reinterpret_cast<const uint4*>(&lds_in[(ry * 18 + rxc) * 64 + chb]);
            bf[n].u[0] = bp[0]; bf[n].u[1] = bp[1];
        }
        #pragma unroll
        for (int n = 0; n < 4; ++n) {
            acc[0][n] = __builtin_amdgcn_wmma_f32_16x16x32_bf16(
                false, a[0].v, false, bf[n].v, (short)0, acc[0][n], false, false);
            acc[1][n] = __builtin_amdgcn_wmma_f32_16x16x32_bf16(
                false, a[1].v, false, bf[n].v, (short)0, acc[1][n], false, false);
        }
    }

    if (MODE == 0) {
        #pragma unroll
        for (int j = 0; j < 2; ++j) {
            const int ocb = (m0 + j) * 16 + hi * 8;   // 8 consecutive oc per lane
            #pragma unroll
            for (int n = 0; n < 4; ++n) {
                int row = nbase + n;
                union { unsigned short s[8]; uint4 u; } pk;
                #pragma unroll
                for (int r = 0; r < 8; ++r) {
                    float v = acc[j][n][r] + shift[ocb + r];
                    pk.s[r] = f2bf(v > 0.f ? v : 0.f);
                }
                *reinterpret_cast<uint4*>(
                    out + (((size_t)(b * 384 + y0 + row) * 384 + (x0 + col)) * 64 + ocb)) = pk.u;
            }
        }
    } else {
        #pragma unroll
        for (int n = 0; n < 4; ++n) {
            int row = nbase + n;
            float mx = -3.4e38f;
            #pragma unroll
            for (int j = 0; j < 2; ++j) {
                const int ocb = (m0 + j) * 16 + hi * 8;
                #pragma unroll
                for (int r = 0; r < 8; ++r) {
                    float v = acc[j][n][r] + shift[ocb + r];
                    mx = v > mx ? v : mx;
                }
            }
            float other = __shfl_xor(mx, 16, 32);   // combine channel halves (lane ^ 16)
            mx = other > mx ? other : mx;
            if (!hi) atomicMax(&lds_max[row * 16 + col], fkey(mx));  // combine M-waves
        }
        __syncthreads();
        {
            int row = tid >> 4, c2 = tid & 15;
            float v = funkey(lds_max[tid]);
            sa[((size_t)b * 384 + (y0 + row)) * 384 + (x0 + c2)] = 1.f / (1.f + __expf(-v));
        }
    }
}

// ---------- vertical window sums: V[y][x] = sum_{dy<cs} sa[y+dy][x], y in [0,Hc) ----------
__global__ void colwin(const float* __restrict__ sa, float* __restrict__ V, const int* csp) {
    int t = blockIdx.x * 256 + threadIdx.x;
    if (t >= 8 * 384) return;
    int b = t / 384, xc = t % 384;
    int cs = *csp, Hc = 384 - cs;
    const float* s = sa + (size_t)b * 384 * 384 + xc;
    float* vp = V + (size_t)b * 384 * 384 + xc;
    float run = 0.f;
    for (int y = 0; y < cs; ++y) run += s[(size_t)y * 384];
    vp[0] = run;
    for (int y = 1; y < Hc; ++y) {
        run += s[(size_t)(y + cs - 1) * 384] - s[(size_t)(y - 1) * 384];
        vp[(size_t)y * 384] = run;
    }
}

// ---------- horizontal window + per-row argmax ----------
__global__ void rowscan(const float* __restrict__ V, uint2* __restrict__ rowred, const int* csp) {
    int t = blockIdx.x * 256 + threadIdx.x;
    if (t >= 8 * 384) return;
    int b = t / 384, y = t % 384;
    int cs = *csp, Hc = 384 - cs, Wc = 384 - cs;
    if (y >= Hc) return;
    const float* vp = V + ((size_t)b * 384 + y) * 384;
    float run = 0.f;
    for (int x = 0; x < cs; ++x) run += vp[x];
    float best = run; int bestx = 0;
    for (int x = 1; x < Wc; ++x) {
        run += vp[x + cs - 1] - vp[x - 1];
        if (run > best) { best = run; bestx = x; }   // strict > keeps first occurrence
    }
    rowred[t] = make_uint2(fkey(best), (unsigned)(y * Wc + bestx));
}

// ---------- per-batch argmax reduce ----------
__global__ void argmax_final(const uint2* __restrict__ rowred, int* __restrict__ rc, const int* csp) {
    __shared__ unsigned skey[256]; __shared__ unsigned sidx[256];
    int b = blockIdx.x, tid = threadIdx.x;
    int cs = *csp, Hc = 384 - cs, Wc = 384 - cs;
    unsigned bk = 0u, bi = 0xffffffffu;
    for (int y = tid; y < Hc; y += 256) {
        uint2 e = rowred[b * 384 + y];
        if (e.x > bk || (e.x == bk && e.y < bi)) { bk = e.x; bi = e.y; }
    }
    skey[tid] = bk; sidx[tid] = bi;
    __syncthreads();
    for (int s = 128; s > 0; s >>= 1) {
        if (tid < s) {
            if (skey[tid + s] > skey[tid] ||
                (skey[tid + s] == skey[tid] && sidx[tid + s] < sidx[tid])) {
                skey[tid] = skey[tid + s]; sidx[tid] = sidx[tid + s];
            }
        }
        __syncthreads();
    }
    if (tid == 0) {
        rc[b * 2]     = (int)(sidx[0] / (unsigned)Wc);
        rc[b * 2 + 1] = (int)(sidx[0] % (unsigned)Wc);
    }
}

// ---------- crop gather ----------
__global__ void crop_kernel(const float* __restrict__ x, const int* __restrict__ rc,
                            float* __restrict__ out, const int* csp) {
    int cs = *csp;
    size_t total = (size_t)8 * 3 * cs * cs;
    size_t i = (size_t)blockIdx.x * 256 + threadIdx.x;
    if (i >= total) return;
    int xx = (int)(i % cs); size_t t = i / cs;
    int yy = (int)(t % cs); t /= cs;
    int ch = (int)(t % 3);  int b = (int)(t / 3);
    int r = rc[b * 2], c = rc[b * 2 + 1];
    out[i] = x[(((size_t)b * 3 + ch) * 384 + (r + yy)) * 384 + (c + xx)];
}

// ---------- launch ----------
extern "C" void kernel_launch(void* const* d_in, const int* in_sizes, int n_in,
                              void* d_out, int out_size, void* d_ws, size_t ws_size,
                              hipStream_t stream) {
    const float* x  = (const float*)d_in[0];
    const float* w1 = (const float*)d_in[1];
    const float* w2 = (const float*)d_in[2];
    const float* w3 = (const float*)d_in[3];
    const float* g1 = (const float*)d_in[4],  *b1 = (const float*)d_in[5];
    const float* m1 = (const float*)d_in[6],  *v1 = (const float*)d_in[7];
    const float* g2 = (const float*)d_in[8],  *b2 = (const float*)d_in[9];
    const float* m2 = (const float*)d_in[10], *v2 = (const float*)d_in[11];
    const float* g3 = (const float*)d_in[12], *b3 = (const float*)d_in[13];
    const float* m3 = (const float*)d_in[14], *v3 = (const float*)d_in[15];
    const int* csp  = (const int*)d_in[16];
    float* out = (float*)d_out;
    (void)in_sizes; (void)n_in; (void)ws_size;

    char* ws = (char*)d_ws;
    size_t o = 0;
    auto take = [&](size_t bytes) -> char* {
        char* p = ws + o; o += (bytes + 255) & ~(size_t)255; return p;
    };
    unsigned short* h1  = (unsigned short*)take((size_t)8 * 384 * 384 * 64 * 2);
    unsigned short* h2  = (unsigned short*)take((size_t)8 * 384 * 384 * 64 * 2);
    unsigned short* wA2 = (unsigned short*)take((size_t)4 * 18 * 32 * 16 * 2);
    unsigned short* wA3 = (unsigned short*)take((size_t)4 * 18 * 32 * 16 * 2);
    float* w1s      = (float*)take(64 * 27 * 4);
    float* scaleAll = (float*)take(192 * 4);
    float* shiftAll = (float*)take(192 * 4);
    float* sa       = (float*)take((size_t)8 * 384 * 384 * 4);
    float* V        = (float*)take((size_t)8 * 384 * 384 * 4);
    uint2* rowred   = (uint2*)take((size_t)8 * 384 * 8);
    int*   rc       = (int*)take(64);

    prep_bn<<<1, 192, 0, stream>>>(g1, b1, m1, v1, g2, b2, m2, v2, g3, b3, m3, v3,
                                   scaleAll, shiftAll);
    prep_w1<<<7, 256, 0, stream>>>(w1, scaleAll, w1s);
    prep_wA<<<144, 256, 0, stream>>>(w2, scaleAll + 64,  wA2);
    prep_wA<<<144, 256, 0, stream>>>(w3, scaleAll + 128, wA3);

    conv1_kernel<<<18432, 256, 0, stream>>>(x, w1s, shiftAll, h1);
    conv64_wmma<0><<<dim3(24, 24, 8), 256, 0, stream>>>(h1, wA2, shiftAll + 64, h2, nullptr);
    conv64_wmma<1><<<dim3(24, 24, 8), 256, 0, stream>>>(h2, wA3, shiftAll + 128, nullptr, sa);

    colwin<<<12, 256, 0, stream>>>(sa, V, csp);
    rowscan<<<12, 256, 0, stream>>>(V, rowred, csp);
    argmax_final<<<8, 256, 0, stream>>>(rowred, rc, csp);
    crop_kernel<<<(out_size + 255) / 256, 256, 0, stream>>>(x, rc, out, csp);
}
